// AxisAwareGPTWithMoEImproved_62766652064418
// MI455X (gfx1250) — compile-verified, matrix operand order checked by reading, not run on hardware
//
#include <hip/hip_runtime.h>

// ---------------------------------------------------------------------------
// Types for WMMA (gfx1250, wave32)
// ---------------------------------------------------------------------------
typedef unsigned short u16;
typedef __attribute__((ext_vector_type(16))) __bf16        bf16x16;
typedef __attribute__((ext_vector_type(8)))  float         f32x8;
typedef __attribute__((ext_vector_type(4)))  unsigned int  u32x4;

union FragU { bf16x16 v; u32x4 q[2]; };

__device__ __forceinline__ u16 f2bf(float f) {
    unsigned u = __float_as_uint(f);
    unsigned r = (u + 0x7FFFu + ((u >> 16) & 1u)) >> 16;
    return (u16)r;
}

// Model dims
#define Bsz 4
#define Tsz 512
#define Dm  512
#define NHd 8
#define HDd 64
#define DSd 128
#define NEe 8
#define DAa 256
#define DFFf 2048
#define EFFf 512
#define Vv  32000
#define NAXx 12
#define MT  (Bsz*Tsz)   // 2048 tokens

// Output tuple offsets (floats), in reference return order
#define O_LOGITS 0L
#define O_AXIS   65536000L
#define O_TRAJ   (O_AXIS + 24576L)
#define O_GL     (O_TRAJ + 262144L)
#define O_SL     (O_GL + 1L)
#define O_UNC    (O_SL + 1L)
#define O_TEMP   (O_UNC + 2048L)
#define O_INF    (O_TEMP + 2048L)

enum { GF_RELU = 1, GF_RES = 2, GF_F32 = 4, GF_B16 = 8 };

// ---------------------------------------------------------------------------
// Generic batched WMMA GEMM: D[m,n] = sum_k A[m,k] * Wt[n,k]  (+bias,+res)
// A: bf16 [.., M, K] row-major (lda), batch offsets b*sAb + h*sAh
// Wt: bf16 [.., N, K] row-major (ldb) i.e. weight pre-transposed
// Each wave computes a 16x64 strip; K in steps of 32 via v_wmma_f32_16x16x32_bf16
// ---------------------------------------------------------------------------
__global__ void k_gemm(const u16* __restrict__ A, long lda, long sAb, long sAh,
                       const u16* __restrict__ Bt, long ldb, long sBb, long sBh,
                       float* __restrict__ C, long ldc, long sCb, long sCh,
                       const float* __restrict__ bias, long sBiasB,
                       const float* __restrict__ R,
                       u16* __restrict__ C16,
                       int M, int N, int K, int nb, int nh, int flags)
{
    int wave = (int)((blockIdx.x * blockDim.x + threadIdx.x) >> 5);
    int lane = (int)(threadIdx.x & 31);
    int Mt = M >> 4, Ns = N >> 6;
    int perBatch = Mt * Ns;
    int total = nb * nh * perBatch;
    if (wave >= total) return;

    int batch = wave / perBatch;
    int t     = wave - batch * perBatch;
    int b     = batch / nh;
    int h     = batch - b * nh;
    int mt    = t / Ns;
    int ns    = t - mt * Ns;
    int m0 = mt << 4, n0 = ns << 6;

    int r  = lane & 15;
    int hi = lane >> 4;     // which K-half this lane holds

    const u16* arow = A + (long)b * sAb + (long)h * sAh + (long)(m0 + r) * lda;
    const u16* brow0 = Bt + (long)b * sBb + (long)h * sBh + (long)(n0 +  0 + r) * ldb;
    const u16* brow1 = Bt + (long)b * sBb + (long)h * sBh + (long)(n0 + 16 + r) * ldb;
    const u16* brow2 = Bt + (long)b * sBb + (long)h * sBh + (long)(n0 + 32 + r) * ldb;
    const u16* brow3 = Bt + (long)b * sBb + (long)h * sBh + (long)(n0 + 48 + r) * ldb;

    f32x8 acc0 = {}, acc1 = {}, acc2 = {}, acc3 = {};

    for (int k0 = 0; k0 < K; k0 += 32) {
        FragU af;
        af.q[0] = *(const u32x4*)(arow + k0      + hi * 8);
        af.q[1] = *(const u32x4*)(arow + k0 + 16 + hi * 8);

        FragU b0, b1, b2, b3;
        b0.q[0] = *(const u32x4*)(brow0 + k0      + hi * 8);
        b0.q[1] = *(const u32x4*)(brow0 + k0 + 16 + hi * 8);
        b1.q[0] = *(const u32x4*)(brow1 + k0      + hi * 8);
        b1.q[1] = *(const u32x4*)(brow1 + k0 + 16 + hi * 8);
        b2.q[0] = *(const u32x4*)(brow2 + k0      + hi * 8);
        b2.q[1] = *(const u32x4*)(brow2 + k0 + 16 + hi * 8);
        b3.q[0] = *(const u32x4*)(brow3 + k0      + hi * 8);
        b3.q[1] = *(const u32x4*)(brow3 + k0 + 16 + hi * 8);

        acc0 = __builtin_amdgcn_wmma_f32_16x16x32_bf16(false, af.v, false, b0.v,
                                                       (short)0, acc0, false, false);
        acc1 = __builtin_amdgcn_wmma_f32_16x16x32_bf16(false, af.v, false, b1.v,
                                                       (short)0, acc1, false, false);
        acc2 = __builtin_amdgcn_wmma_f32_16x16x32_bf16(false, af.v, false, b2.v,
                                                       (short)0, acc2, false, false);
        acc3 = __builtin_amdgcn_wmma_f32_16x16x32_bf16(false, af.v, false, b3.v,
                                                       (short)0, acc3, false, false);
    }

    // Epilogue. C/D layout: lane 0-15 -> N=lane, VGPR rr -> M=rr (+8 for lanes 16-31)
    long cbase = (long)b * sCb + (long)h * sCh;
    int cm0 = m0 + hi * 8;
    f32x8 accs[4] = { acc0, acc1, acc2, acc3 };
    #pragma unroll
    for (int i = 0; i < 4; i++) {
        int n = n0 + i * 16 + r;
        float bv = bias ? bias[(long)b * sBiasB + n] : 0.f;
        #pragma unroll
        for (int rr = 0; rr < 8; rr++) {
            long cidx = cbase + (long)(cm0 + rr) * ldc + n;
            float v = accs[i][rr] + bv;
            if (flags & GF_RELU) v = fmaxf(v, 0.f);
            if (flags & GF_RES)  v += R[cidx];
            if (flags & GF_F32)  C[cidx]   = v;
            if (flags & GF_B16)  C16[cidx] = f2bf(v);
        }
    }
}

// ---------------------------------------------------------------------------
// Weight pack: W (K x N, f32 row-major) -> Wt (N x K, bf16 row-major)
// ---------------------------------------------------------------------------
__global__ void k_pack(const float* __restrict__ W, u16* __restrict__ Wt, int K, int N)
{
    long i = (long)blockIdx.x * blockDim.x + threadIdx.x;
    long total = (long)K * N;
    if (i >= total) return;
    long n = i / K, k = i - n * K;
    Wt[i] = f2bf(W[k * (long)N + n]);
}

// ---------------------------------------------------------------------------
// Per-batch axis vectors: a = axis_emb[axis_id[b]]; gamma/beta/ag/abias
// ---------------------------------------------------------------------------
__global__ void k_axis(const int* __restrict__ axis_id, const float* __restrict__ axis_emb,
                       const float* __restrict__ fgw, const float* __restrict__ fgb,
                       const float* __restrict__ fbw, const float* __restrict__ fbb,
                       const float* __restrict__ wag,
                       const float* __restrict__ wa,  const float* __restrict__ sb,
                       float* __restrict__ gamma, float* __restrict__ beta,
                       float* __restrict__ ag, float* __restrict__ abias)
{
    __shared__ float a[DAa];
    int b = blockIdx.x, tid = threadIdx.x;   // 256 threads
    a[tid] = axis_emb[(long)axis_id[b] * DAa + tid];
    __syncthreads();
    for (int d = tid; d < Dm; d += 256) {
        float g = fgb[d], be = fbb[d];
        for (int i = 0; i < DAa; i++) {
            g  += a[i] * fgw[(long)i * Dm + d];
            be += a[i] * fbw[(long)i * Dm + d];
        }
        gamma[b * Dm + d] = g;
        beta[b * Dm + d]  = be;
    }
    if (tid < NEe) {
        float s = 0.f;
        for (int i = 0; i < DAa; i++) s += a[i] * wag[i * NEe + tid];
        ag[b * NEe + tid] = s;
    }
    if (tid < DSd) {
        float s = sb[tid];
        for (int i = 0; i < DAa; i++) s += a[i] * wa[i * DSd + tid];
        abias[b * DSd + tid] = s;
    }
}

// Embedding gather + FiLM
__global__ void k_embed(const int* __restrict__ tokens, const float* __restrict__ emb,
                        const float* __restrict__ gamma, const float* __restrict__ beta,
                        float* __restrict__ x32, u16* __restrict__ x16)
{
    int idx = blockIdx.x * blockDim.x + threadIdx.x;  // MT * Dm
    int d = idx & (Dm - 1);
    int bt = idx >> 9;
    int b = bt >> 9;
    int tok = tokens[bt];
    float v = gamma[b * Dm + d] * emb[(long)tok * Dm + d] + beta[b * Dm + d];
    x32[idx] = v;
    x16[idx] = f2bf(v);
}

// Row LayerNorm (width 512), writes f32 + bf16
__global__ void k_ln(const float* __restrict__ x, const float* __restrict__ g,
                     const float* __restrict__ b, float* __restrict__ y32,
                     u16* __restrict__ y16)
{
    __shared__ float red[256];
    __shared__ float mv[2];
    int m = blockIdx.x, tid = threadIdx.x;
    float v0 = x[(long)m * Dm + tid], v1 = x[(long)m * Dm + 256 + tid];
    red[tid] = v0 + v1; __syncthreads();
    for (int o = 128; o; o >>= 1) { if (tid < o) red[tid] += red[tid + o]; __syncthreads(); }
    if (tid == 0) mv[0] = red[0] / 512.f;
    __syncthreads();
    float mean = mv[0];
    float d0 = v0 - mean, d1 = v1 - mean;
    red[tid] = d0 * d0 + d1 * d1; __syncthreads();
    for (int o = 128; o; o >>= 1) { if (tid < o) red[tid] += red[tid + o]; __syncthreads(); }
    if (tid == 0) mv[1] = rsqrtf(red[0] / 512.f + 1e-5f);
    __syncthreads();
    float rs = mv[1];
    float o0 = d0 * rs * g[tid] + b[tid];
    float o1 = d1 * rs * g[256 + tid] + b[256 + tid];
    y32[(long)m * Dm + tid] = o0;        y16[(long)m * Dm + tid] = f2bf(o0);
    y32[(long)m * Dm + 256 + tid] = o1;  y16[(long)m * Dm + 256 + tid] = f2bf(o1);
}

// Softmax over rows of 512 (scaled by 1/sqrt(64)), f32 in -> bf16 probs out
__global__ void k_softmax(const float* __restrict__ scores, u16* __restrict__ probs)
{
    __shared__ float red[256];
    __shared__ float bc;
    long row = blockIdx.x;
    int tid = threadIdx.x;
    const float* p = scores + row * 512;
    float v0 = p[tid] * 0.125f, v1 = p[256 + tid] * 0.125f;
    red[tid] = fmaxf(v0, v1); __syncthreads();
    for (int o = 128; o; o >>= 1) { if (tid < o) red[tid] = fmaxf(red[tid], red[tid + o]); __syncthreads(); }
    if (!tid) bc = red[0];
    __syncthreads();
    float mx = bc;
    float e0 = expf(v0 - mx), e1 = expf(v1 - mx);
    red[tid] = e0 + e1; __syncthreads();
    for (int o = 128; o; o >>= 1) { if (tid < o) red[tid] += red[tid + o]; __syncthreads(); }
    if (!tid) bc = red[0];
    __syncthreads();
    float inv = 1.f / bc;
    probs[row * 512 + tid]       = f2bf(e0 * inv);
    probs[row * 512 + 256 + tid] = f2bf(e1 * inv);
}

// Pack V^T per (b,h): vt[b,h,d,t] = qkv16[b,t, 1024 + h*64 + d]
__global__ void k_pack_vt(const u16* __restrict__ qkv16, u16* __restrict__ vt)
{
    int idx = blockIdx.x * blockDim.x + threadIdx.x;  // Bsz*NHd*HDd*Tsz
    int t = idx & 511;
    int rr = idx >> 9;
    int d = rr & 63; rr >>= 6;
    int h = rr & 7;
    int b = rr >> 3;
    vt[idx] = qkv16[((long)(b * Tsz + t)) * (3 * Dm) + 2 * Dm + h * HDd + d];
}

// Gating: logits = h @ wg + bg + ag[b]; softmax over 8
__global__ void k_gate(const float* __restrict__ h, const float* __restrict__ wg,
                       const float* __restrict__ bg, const float* __restrict__ ag,
                       float* __restrict__ gate)
{
    __shared__ float red[128];
    __shared__ float lg[NEe];
    int m = blockIdx.x, tid = threadIdx.x;  // 128 threads
    int e = tid >> 4, li = tid & 15;
    float s = 0.f;
    for (int d = li; d < Dm; d += 16) s += h[(long)m * Dm + d] * wg[d * NEe + e];
    red[tid] = s; __syncthreads();
    for (int o = 8; o; o >>= 1) { if (li < o) red[tid] += red[tid + o]; __syncthreads(); }
    if (li == 0) lg[e] = red[tid] + bg[e] + ag[(m >> 9) * NEe + e];
    __syncthreads();
    if (tid == 0) {
        float mx = -1e30f;
        for (int i = 0; i < NEe; i++) mx = fmaxf(mx, lg[i]);
        float sum = 0.f, ex[NEe];
        for (int i = 0; i < NEe; i++) { ex[i] = expf(lg[i] - mx); sum += ex[i]; }
        for (int i = 0; i < NEe; i++) gate[(long)m * NEe + i] = ex[i] / sum;
    }
}

// h2 = h + sum_e gate[m,e]*eo[e,m,d]
__global__ void k_moe_combine(const float* __restrict__ h, const float* __restrict__ gate,
                              const float* __restrict__ eo, float* __restrict__ h2,
                              u16* __restrict__ h2b)
{
    int idx = blockIdx.x * blockDim.x + threadIdx.x;  // MT*Dm
    int m = idx >> 9;
    float s = h[idx];
    #pragma unroll
    for (int e = 0; e < NEe; e++)
        s += gate[(long)m * NEe + e] * eo[(long)e * MT * Dm + idx];
    h2[idx] = s;
    h2b[idx] = f2bf(s);
}

// gate_loss and stability_loss (single block)
__global__ void k_gate_losses(const float* __restrict__ gate, float* __restrict__ out_gl,
                              float* __restrict__ out_sl)
{
    __shared__ float red[256];
    int tid = threadIdx.x;
    float gl = 0.f;
    for (int e = 0; e < NEe; e++) {
        float s = 0.f;
        for (int m = tid; m < MT; m += 256) s += gate[(long)m * NEe + e];
        red[tid] = s; __syncthreads();
        for (int o = 128; o; o >>= 1) { if (tid < o) red[tid] += red[tid + o]; __syncthreads(); }
        if (tid == 0) { float imp = red[0] / (float)MT; gl += imp * imp; }
        __syncthreads();
    }
    if (tid == 0) *out_gl = (float)NEe * gl;
    float s = 0.f;
    for (int idx = tid; idx < Bsz * (Tsz - 1) * NEe; idx += 256) {
        int e = idx % NEe;
        int rr = idx / NEe;
        int b = rr / (Tsz - 1);
        int t = rr % (Tsz - 1) + 1;
        int m = b * Tsz + t;
        float d = gate[(long)m * NEe + e] - gate[(long)(m - 1) * NEe + e];
        s += d * d;
    }
    red[tid] = s; __syncthreads();
    for (int o = 128; o; o >>= 1) { if (tid < o) red[tid] += red[tid + o]; __syncthreads(); }
    if (tid == 0) *out_sl = red[0] / ((float)Bsz * (Tsz - 1) * NEe);
}

// SSM recurrence: s' = tanh(s@A + hB[b,t] + abias[b]); traj -> d_out (f32) + bf16
// One block per b, 512 threads, A cached in dynamic LDS (64KB)
__global__ void k_ssm(const float* __restrict__ A, const float* __restrict__ hB,
                      const float* __restrict__ abias, float* __restrict__ traj_out,
                      u16* __restrict__ traj16)
{
    extern __shared__ float sm[];
    float* As  = sm;                 // 128*128
    float* s   = sm + DSd * DSd;     // 128
    float* red = s + DSd;            // 512
    int b = blockIdx.x, tid = threadIdx.x;
    for (int i = tid; i < DSd * DSd; i += 512) As[i] = A[i];
    if (tid < DSd) s[tid] = 0.f;
    __syncthreads();
    int j = tid & 127, seg = tid >> 7;
    for (int t = 0; t < Tsz; t++) {
        float p = 0.f;
        for (int i = seg * 32; i < seg * 32 + 32; i++) p += s[i] * As[i * DSd + j];
        red[tid] = p; __syncthreads();
        if (tid < DSd) {
            float v = red[j] + red[128 + j] + red[256 + j] + red[384 + j]
                    + hB[(long)(b * Tsz + t) * DSd + j] + abias[b * DSd + j];
            v = tanhf(v);
            s[j] = v;
            traj_out[(long)(b * Tsz + t) * DSd + j] = v;
            traj16[(long)(b * Tsz + t) * DSd + j]  = f2bf(v);
        }
        __syncthreads();
    }
}

// Small heads: axis_pred(12), inferred(12), unc(1), temp(1) per token
__global__ void k_heads(const float* __restrict__ comb,
                        const float* __restrict__ axw, const float* __restrict__ axb,
                        const float* __restrict__ aiw, const float* __restrict__ aib,
                        const float* __restrict__ uw,  const float* __restrict__ ub,
                        const float* __restrict__ tw,  const float* __restrict__ tb,
                        float* __restrict__ out_axis, float* __restrict__ out_inf,
                        float* __restrict__ out_unc,  float* __restrict__ out_temp)
{
    __shared__ float row[Dm];
    int m = blockIdx.x, tid = threadIdx.x;  // 64 threads
    for (int i = tid; i < Dm; i += 64) row[i] = comb[(long)m * Dm + i];
    __syncthreads();
    if (tid < NAXx) {
        float s = axb[tid];
        for (int d = 0; d < Dm; d++) s += row[d] * axw[d * NAXx + tid];
        out_axis[(long)m * NAXx + tid] = s;
    } else if (tid < 2 * NAXx) {
        int jj = tid - NAXx;
        float s = aib[jj];
        for (int d = 0; d < Dm; d++) s += row[d] * aiw[d * NAXx + jj];
        out_inf[(long)m * NAXx + jj] = s;
    } else if (tid == 24) {
        float s = ub[0];
        for (int d = 0; d < Dm; d++) s += row[d] * uw[d];
        out_unc[m] = s;
    } else if (tid == 25) {
        float s = tb[0];
        for (int d = 0; d < Dm; d++) s += row[d] * tw[d];
        out_temp[m] = 1.f / (1.f + expf(-s)) * 2.f + 0.5f;
    }
}

// ---------------------------------------------------------------------------
// Host launcher
// ---------------------------------------------------------------------------
static inline void gemm(hipStream_t st,
                        const u16* A, long lda, long sAb, long sAh,
                        const u16* Bt, long ldb, long sBb, long sBh,
                        float* C, long ldc, long sCb, long sCh,
                        const float* bias, long sBiasB,
                        const float* R, u16* C16,
                        int M, int N, int K, int nb, int nh, int flags)
{
    long waves = (long)nb * nh * (M / 16) * (N / 64);
    int blocks = (int)((waves + 7) / 8);
    k_gemm<<<blocks, 256, 0, st>>>(A, lda, sAb, sAh, Bt, ldb, sBb, sBh,
                                   C, ldc, sCb, sCh, bias, sBiasB, R, C16,
                                   M, N, K, nb, nh, flags);
}

extern "C" void kernel_launch(void* const* d_in, const int* in_sizes, int n_in,
                              void* d_out, int out_size, void* d_ws, size_t ws_size,
                              hipStream_t stream)
{
    (void)in_sizes; (void)n_in; (void)out_size; (void)ws_size;

    const int*   TOKENS  = (const int*)d_in[0];
    const int*   AXIS_ID = (const int*)d_in[1];
    const float* TOK_EMB = (const float*)d_in[2];
    const float* AX_EMB  = (const float*)d_in[3];
    const float* FGW = (const float*)d_in[4];
    const float* FGB = (const float*)d_in[5];
    const float* FBW = (const float*)d_in[6];
    const float* FBB = (const float*)d_in[7];
    const float* MOE_WG = (const float*)d_in[8];
    const float* MOE_BG = (const float*)d_in[9];
    const float* MOE_WAG = (const float*)d_in[10];
    const float* MOE_W1 = (const float*)d_in[11];
    const float* MOE_B1 = (const float*)d_in[12];
    const float* MOE_W2 = (const float*)d_in[13];
    const float* MOE_B2 = (const float*)d_in[14];
    const float* SSM_A  = (const float*)d_in[15];
    const float* SSM_Bw = (const float*)d_in[16];
    const float* SSM_b  = (const float*)d_in[17];
    const float* SSM_WA = (const float*)d_in[18];
    const float* SSM_C  = (const float*)d_in[19];
    const float* SSM_CB = (const float*)d_in[20];
    const float* LM_W   = (const float*)d_in[21];
    const float* LM_B   = (const float*)d_in[22];
    const float* AXW = (const float*)d_in[23];
    const float* AXB = (const float*)d_in[24];
    const float* AIW = (const float*)d_in[25];
    const float* AIB = (const float*)d_in[26];
    const float* UW  = (const float*)d_in[27];
    const float* UB  = (const float*)d_in[28];
    const float* TW  = (const float*)d_in[29];
    const float* TB  = (const float*)d_in[30];

    float* dout = (float*)d_out;

    // ---- workspace carve (bump allocator, deterministic per call) ----
    char* base = (char*)d_ws;
    size_t off = 0;
    auto carve = [&](size_t bytes) -> void* {
        void* p = base + off;
        off += (bytes + 255) & ~(size_t)255;
        return p;
    };

    u16* wqkvt[4]; u16* wot[4]; u16* w1t[4]; u16* w2t[4];
    for (int l = 0; l < 4; l++) {
        wqkvt[l] = (u16*)carve((size_t)Dm * 3 * Dm * 2);
        wot[l]   = (u16*)carve((size_t)Dm * Dm * 2);
        w1t[l]   = (u16*)carve((size_t)Dm * DFFf * 2);
        w2t[l]   = (u16*)carve((size_t)DFFf * Dm * 2);
    }
    u16* moe1t = (u16*)carve((size_t)NEe * Dm * EFFf * 2);
    u16* moe2t = (u16*)carve((size_t)NEe * EFFf * Dm * 2);
    u16* ssmBt = (u16*)carve((size_t)Dm * DSd * 2);
    u16* ssmCt = (u16*)carve((size_t)DSd * Dm * 2);
    u16* lmwt  = (u16*)carve((size_t)Dm * Vv * 2);

    float* gamma = (float*)carve((size_t)Bsz * Dm * 4);
    float* beta  = (float*)carve((size_t)Bsz * Dm * 4);
    float* ag    = (float*)carve((size_t)Bsz * NEe * 4);
    float* abias = (float*)carve((size_t)Bsz * DSd * 4);

    float* x32   = (float*)carve((size_t)MT * Dm * 4);
    u16*   x16   = (u16*)  carve((size_t)MT * Dm * 2);
    float* tmp32 = (float*)carve((size_t)MT * Dm * 4);
    u16*   qkv16 = (u16*)  carve((size_t)MT * 3 * Dm * 2);
    float* scor  = (float*)carve((size_t)Bsz * NHd * Tsz * Tsz * 4);
    u16*   prob  = (u16*)  carve((size_t)Bsz * NHd * Tsz * Tsz * 2);
    u16*   vt16  = (u16*)  carve((size_t)Bsz * NHd * HDd * Tsz * 2);
    u16*   o16   = (u16*)  carve((size_t)MT * Dm * 2);
    u16*   ff16  = (u16*)  carve((size_t)MT * DFFf * 2);
    float* gate  = (float*)carve((size_t)MT * NEe * 4);
    u16*   eh16  = (u16*)  carve((size_t)NEe * MT * EFFf * 2);
    float* eo32  = (float*)carve((size_t)NEe * MT * Dm * 4);
    float* h2_32 = (float*)carve((size_t)MT * Dm * 4);
    u16*   h2_16 = (u16*)  carve((size_t)MT * Dm * 2);
    float* hB32  = (float*)carve((size_t)MT * DSd * 4);
    u16*   traj16= (u16*)  carve((size_t)MT * DSd * 2);
    float* comb32= (float*)carve((size_t)MT * Dm * 4);
    u16*   comb16= (u16*)  carve((size_t)MT * Dm * 2);

    auto pack = [&](const float* W, u16* Wt, long K, long N) {
        long total = K * N;
        int blocks = (int)((total + 255) / 256);
        k_pack<<<blocks, 256, 0, stream>>>(W, Wt, (int)K, (int)N);
    };

    // ---- pack all weights to transposed bf16 ----
    for (int l = 0; l < 4; l++) {
        const float* WQKV = (const float*)d_in[31 + l * 12 + 0];
        const float* WO   = (const float*)d_in[31 + l * 12 + 2];
        const float* W1   = (const float*)d_in[31 + l * 12 + 6];
        const float* W2   = (const float*)d_in[31 + l * 12 + 8];
        pack(WQKV, wqkvt[l], Dm, 3 * Dm);
        pack(WO,   wot[l],   Dm, Dm);
        pack(W1,   w1t[l],   Dm, DFFf);
        pack(W2,   w2t[l],   DFFf, Dm);
    }
    for (int e = 0; e < NEe; e++) {
        pack(MOE_W1 + (long)e * Dm * EFFf, moe1t + (long)e * Dm * EFFf, Dm, EFFf);
        pack(MOE_W2 + (long)e * EFFf * Dm, moe2t + (long)e * EFFf * Dm, EFFf, Dm);
    }
    pack(SSM_Bw, ssmBt, Dm, DSd);
    pack(SSM_C,  ssmCt, DSd, Dm);
    pack(LM_W,   lmwt,  Dm, Vv);

    // ---- FiLM + embedding ----
    k_axis<<<Bsz, 256, 0, stream>>>(AXIS_ID, AX_EMB, FGW, FGB, FBW, FBB, MOE_WAG,
                                    SSM_WA, SSM_b, gamma, beta, ag, abias);
    k_embed<<<(MT * Dm) / 256, 256, 0, stream>>>(TOKENS, TOK_EMB, gamma, beta, x32, x16);

    // ---- transformer layers ----
    for (int l = 0; l < 4; l++) {
        const float* BQKV = (const float*)d_in[31 + l * 12 + 1];
        const float* BO   = (const float*)d_in[31 + l * 12 + 3];
        const float* LN1G = (const float*)d_in[31 + l * 12 + 4];
        const float* LN1B = (const float*)d_in[31 + l * 12 + 5];
        const float* B1   = (const float*)d_in[31 + l * 12 + 7];
        const float* B2   = (const float*)d_in[31 + l * 12 + 9];
        const float* LN2G = (const float*)d_in[31 + l * 12 + 10];
        const float* LN2B = (const float*)d_in[31 + l * 12 + 11];

        // qkv (bf16 only)
        gemm(stream, x16, Dm, 0, 0, wqkvt[l], Dm, 0, 0,
             nullptr, 3 * Dm, 0, 0, BQKV, 0, nullptr, qkv16,
             MT, 3 * Dm, Dm, 1, 1, GF_B16);
        // scores per (b,h): Q @ K^T
        gemm(stream, qkv16, 3 * Dm, (long)Tsz * 3 * Dm, HDd,
             qkv16 + Dm, 3 * Dm, (long)Tsz * 3 * Dm, HDd,
             scor, Tsz, (long)NHd * Tsz * Tsz, (long)Tsz * Tsz,
             nullptr, 0, nullptr, nullptr, Tsz, Tsz, HDd, Bsz, NHd, GF_F32);
        k_softmax<<<Bsz * NHd * Tsz, 256, 0, stream>>>(scor, prob);
        k_pack_vt<<<(Bsz * NHd * HDd * Tsz) / 256, 256, 0, stream>>>(qkv16, vt16);
        // attn @ V  -> o[b,q,h,d] (bf16 only)
        gemm(stream, prob, Tsz, (long)NHd * Tsz * Tsz, (long)Tsz * Tsz,
             vt16, Tsz, (long)NHd * HDd * Tsz, (long)HDd * Tsz,
             nullptr, Dm, (long)Tsz * Dm, HDd,
             nullptr, 0, nullptr, o16, Tsz, HDd, Tsz, Bsz, NHd, GF_B16);
        // o-proj + residual
        gemm(stream, o16, Dm, 0, 0, wot[l], Dm, 0, 0,
             tmp32, Dm, 0, 0, BO, 0, x32, nullptr,
             MT, Dm, Dm, 1, 1, GF_RES | GF_F32);
        k_ln<<<MT, 256, 0, stream>>>(tmp32, LN1G, LN1B, x32, x16);
        // ff1 (relu, bf16 only)
        gemm(stream, x16, Dm, 0, 0, w1t[l], Dm, 0, 0,
             nullptr, DFFf, 0, 0, B1, 0, nullptr, ff16,
             MT, DFFf, Dm, 1, 1, GF_RELU | GF_B16);
        // ff2 + residual
        gemm(stream, ff16, DFFf, 0, 0, w2t[l], DFFf, 0, 0,
             tmp32, Dm, 0, 0, B2, 0, x32, nullptr,
             MT, Dm, DFFf, 1, 1, GF_RES | GF_F32);
        k_ln<<<MT, 256, 0, stream>>>(tmp32, LN2G, LN2B, x32, x16);
    }

    // ---- MoE ----
    k_gate<<<MT, 128, 0, stream>>>(x32, MOE_WG, MOE_BG, ag, gate);
    gemm(stream, x16, Dm, 0, 0, moe1t, Dm, (long)Dm * EFFf, 0,
         nullptr, EFFf, (long)MT * EFFf, 0, MOE_B1, EFFf, nullptr, eh16,
         MT, EFFf, Dm, NEe, 1, GF_RELU | GF_B16);
    gemm(stream, eh16, EFFf, (long)MT * EFFf, 0, moe2t, EFFf, (long)EFFf * Dm, 0,
         eo32, Dm, (long)MT * Dm, 0, MOE_B2, Dm, nullptr, nullptr,
         MT, Dm, EFFf, NEe, 1, GF_F32);
    k_moe_combine<<<(MT * Dm) / 256, 256, 0, stream>>>(x32, gate, eo32, h2_32, h2_16);
    k_gate_losses<<<1, 256, 0, stream>>>(gate, dout + O_GL, dout + O_SL);

    // ---- SSM ----
    gemm(stream, h2_16, Dm, 0, 0, ssmBt, Dm, 0, 0,
         hB32, DSd, 0, 0, nullptr, 0, nullptr, nullptr,
         MT, DSd, Dm, 1, 1, GF_F32);
    size_t ssm_shmem = (size_t)(DSd * DSd + DSd + 512) * 4;
    k_ssm<<<Bsz, 512, ssm_shmem, stream>>>(SSM_A, hB32, abias, dout + O_TRAJ, traj16);
    // combined = h2 + traj @ C + cb
    gemm(stream, traj16, DSd, 0, 0, ssmCt, DSd, 0, 0,
         comb32, Dm, 0, 0, SSM_CB, 0, h2_32, comb16,
         MT, Dm, DSd, 1, 1, GF_RES | GF_F32 | GF_B16);

    // ---- heads ----
    gemm(stream, comb16, Dm, 0, 0, lmwt, Dm, 0, 0,
         dout + O_LOGITS, Vv, 0, 0, LM_B, 0, nullptr, nullptr,
         MT, Vv, Dm, 1, 1, GF_F32);
    k_heads<<<MT, 64, 0, stream>>>(comb32, AXW, AXB, AIW, AIB, UW, UB, TW, TB,
                                   dout + O_AXIS, dout + O_INF,
                                   dout + O_UNC, dout + O_TEMP);
}
